// Attention_65644280152570
// MI455X (gfx1250) — matrix operand order repaired and checked by reading, not compiled, not run on hardware
//
#include <hip/hip_runtime.h>

typedef __attribute__((ext_vector_type(4)))  unsigned int u32x4;
typedef __attribute__((ext_vector_type(4)))  float        f32x4;
typedef __attribute__((ext_vector_type(16))) __bf16       v16bf;
typedef __attribute__((ext_vector_type(8)))  float        v8f;

union Frag16 { u32x4 q[2]; v16bf v; };

// ---------------- f32 -> bf16 cast, 4 elems / thread ----------------
__global__ void cast_f32_bf16(const float* __restrict__ in, __bf16* __restrict__ out, long n4) {
  long i = (long)blockIdx.x * blockDim.x + threadIdx.x;
  if (i < n4) {
    f32x4 v = ((const f32x4*)in)[i];
    union { __bf16 b[4]; unsigned long long u; } o;
    o.b[0] = (__bf16)v.x; o.b[1] = (__bf16)v.y; o.b[2] = (__bf16)v.z; o.b[3] = (__bf16)v.w;
    ((unsigned long long*)out)[i] = o.u;
  }
}

// ---------------- issue one 128x32 tile load into LDS ----------------
// Fast path: async memory->LDS DMA (2x b128 per thread, ASYNCcnt-tracked).
// 32-bit element offsets throughout (all buffers < 2^31 elements).
__device__ __forceinline__ void issue_tile(__bf16 (*S)[40], const __bf16* __restrict__ G,
                                           int r0, int R, int kk, int sr, int sk, int tid) {
  if (sk == 1) {                       // contiguous along K
    int row = tid >> 1;
    int kh  = (tid & 1) * 16;
    int rg  = r0 + row;
    if (rg < R) {
      // LDS byte address = flat address truncated to 32 bits (aperture rule)
      unsigned lds = (unsigned)(unsigned long long)&S[row][kh];
      const __bf16* g = G + rg * sr + (kk + kh);
      asm volatile("global_load_async_to_lds_b128 %0, %1, off\n\t"
                   "global_load_async_to_lds_b128 %0, %1, off offset:16"
                   :: "v"(lds), "v"(g) : "memory");
    } else {
      u32x4 zz = {0, 0, 0, 0};
      *(u32x4*)&S[row][kh]     = zz;
      *(u32x4*)&S[row][kh + 8] = zz;
    }
  } else if (sr == 1) {                // contiguous along M (transposed operand)
    int k  = tid >> 3;                 // 0..31
    int mc = (tid & 7) * 16;           // 0..112
    const __bf16* p = G + (kk + k) * sk + r0 + mc;
#pragma unroll
    for (int j = 0; j < 16; ++j) {
      __bf16 v = (__bf16)0.0f;
      if (r0 + mc + j < R) v = p[j];
      S[mc + j][k] = v;
    }
  } else {                             // generic gather
    int row = tid >> 1;
    int kh  = (tid & 1) * 16;
    int rg  = r0 + row;
#pragma unroll
    for (int j = 0; j < 16; ++j) {
      __bf16 v = (__bf16)0.0f;
      if (rg < R) v = G[rg * sr + (kk + kh + j) * sk];
      S[row][kh + j] = v;
    }
  }
}

// ---------------- strided-batch WMMA GEMM: C[m,n] = sum_k A(m,k)*B(n,k) ----------------
// Ping-pong LDS tiles: async copies for tile i+1 overlap WMMA compute on tile i.
// Per z-slice: b = z / Hdiv, h = z % Hdiv; operand offsets b*z?b + h*z?h (elements).
template <bool F32OUT>
__global__ __launch_bounds__(256)
void gemm_bt_wmma(const __bf16* __restrict__ A, const __bf16* __restrict__ Bm,
                  void* __restrict__ Cp,
                  int M, int Nn, int K,
                  int sAm, int sAk, int zAb, int zAh,
                  int sBn, int sBk, int zBb, int zBh,
                  int sCm, int sCn, int zCb, int zCh,
                  int Hdiv) {
  __shared__ __bf16 As[2][128][40];
  __shared__ __bf16 Bs[2][128][40];

  int z  = blockIdx.z;
  int bz = z / Hdiv, hz = z % Hdiv;
  const __bf16* Ab = A  + bz * zAb + hz * zAh;
  const __bf16* Bb = Bm + bz * zBb + hz * zBh;
  int cOff = bz * zCb + hz * zCh;

  int tid  = threadIdx.x;
  int wave = tid >> 5;
  int lane = tid & 31;
  int m0   = blockIdx.x * 128;
  int n0   = blockIdx.y * 128;

  v8f acc[8] = {};

  // CDNA5 16-bit fragment addressing (wave32)
  int ml   = wave * 16 + (lane & 15);  // A row within block tile
  int koff = (lane >> 4) * 8;          // A: lanes<16 -> K {0..7,16..23}; >=16 -> {8..15,24..31}
  int kb   = (lane >> 4) * 8;          // B: lanes<16 -> K {0..7,16..23}; >=16 -> {8..15,24..31}
  int nl   = lane & 15;                // B column within 16-wide tile

  const bool bothAsync = (sAk == 1) && (sBk == 1);

  // prologue: stage tile 0 into buffer 0
  issue_tile(As[0], Ab, m0, M,  0, sAm, sAk, tid);
  issue_tile(Bs[0], Bb, n0, Nn, 0, sBn, sBk, tid);

  int p = 0;
  for (int kk = 0; kk < K; kk += 32) {
    if (kk + 32 < K) {
      // prefetch next tile into the alternate buffer (overlaps with compute below)
      issue_tile(As[p ^ 1], Ab, m0, M,  kk + 32, sAm, sAk, tid);
      issue_tile(Bs[p ^ 1], Bb, n0, Nn, kk + 32, sBn, sBk, tid);
      // async loads retire in order: <= (count just issued) outstanding
      // implies the current tile's copies have landed in LDS.
      if (bothAsync) asm volatile("s_wait_asynccnt 0x4" ::: "memory");
      else           asm volatile("s_wait_asynccnt 0x2" ::: "memory");
    } else {
      asm volatile("s_wait_asynccnt 0x0" ::: "memory");
    }
    __syncthreads();

    const __bf16 (*Ac)[40] = As[p];
    const __bf16 (*Bc)[40] = Bs[p];

    Frag16 fa;
    fa.q[0] = *(const u32x4*)&Ac[ml][koff];
    fa.q[1] = *(const u32x4*)&Ac[ml][koff + 16];

    // double-buffered B fragments: load t+1 while WMMA t executes
    Frag16 fb0, fb1;
    fb0.q[0] = *(const u32x4*)&Bc[nl][kb];
    fb0.q[1] = *(const u32x4*)&Bc[nl][kb + 16];
#pragma unroll
    for (int t = 0; t < 8; ++t) {
      Frag16* cur = (t & 1) ? &fb1 : &fb0;
      Frag16* nxt = (t & 1) ? &fb0 : &fb1;
      if (t < 7) {
        nxt->q[0] = *(const u32x4*)&Bc[(t + 1) * 16 + nl][kb];
        nxt->q[1] = *(const u32x4*)&Bc[(t + 1) * 16 + nl][kb + 16];
      }
      acc[t] = __builtin_amdgcn_wmma_f32_16x16x32_bf16(
          false, fa.v, false, cur->v, (short)0, acc[t], false, false);
    }
    __syncthreads();
    p ^= 1;
  }

  int mrow = m0 + wave * 16 + 8 * (lane >> 4);   // C layout: VGPR r -> M = r (+8 for hi lanes)
#pragma unroll
  for (int t = 0; t < 8; ++t) {
    int n = n0 + t * 16 + nl;
    if (n < Nn) {
#pragma unroll
      for (int r = 0; r < 8; ++r) {
        int m = mrow + r;
        if (m < M) {
          int idx = cOff + m * sCm + n * sCn;
          float v = acc[t][r];
          if (F32OUT) ((float*)Cp)[idx] = v;
          else        ((__bf16*)Cp)[idx] = (__bf16)v;
        }
      }
    }
  }
}

// ---------------- per-z sum / sum-of-squares reduction ----------------
__global__ void reduce_stats(const float* __restrict__ x, float* __restrict__ stats,
                             long cntPerZ, long chunk) {
  long z = blockIdx.y;
  const float* p = x + z * cntPerZ + (long)blockIdx.x * chunk;
  float s = 0.f, ss = 0.f;
  for (long i = threadIdx.x; i < chunk; i += blockDim.x) {
    float v = p[i]; s += v; ss += v * v;
  }
  __shared__ float sh[256], sh2[256];
  int tid = threadIdx.x;
  sh[tid] = s; sh2[tid] = ss;
  __syncthreads();
  for (int st = 128; st > 0; st >>= 1) {
    if (tid < st) { sh[tid] += sh[tid + st]; sh2[tid] += sh2[tid + st]; }
    __syncthreads();
  }
  if (tid == 0) {
    atomicAdd(&stats[z * 2],     sh[0]);
    atomicAdd(&stats[z * 2 + 1], sh2[0]);
  }
}

// ---------------- fused instance-norm + row softmax + bf16 cast ----------------
__global__ void norm_softmax(const float* __restrict__ x, __bf16* __restrict__ y,
                             const float* __restrict__ stats,
                             int rowsPerZ, int L, float invCnt) {
  __shared__ float buf[2048];
  __shared__ float red[256];
  long row = blockIdx.x;
  int  z   = (int)(row / rowsPerZ);
  float mean = stats[z * 2] * invCnt;
  float var  = stats[z * 2 + 1] * invCnt - mean * mean;
  float rs   = rsqrtf(var + 1e-5f);
  const float* p = x + row * (long)L;
  int tid = threadIdx.x;

  float mx = -3.4e38f;
  for (int i = tid; i < L; i += 256) {
    float v = (p[i] - mean) * rs;
    buf[i] = v;
    mx = fmaxf(mx, v);
  }
  red[tid] = mx; __syncthreads();
  for (int st = 128; st > 0; st >>= 1) { if (tid < st) red[tid] = fmaxf(red[tid], red[tid + st]); __syncthreads(); }
  mx = red[0]; __syncthreads();

  float s = 0.f;
  for (int i = tid; i < L; i += 256) {
    float e = __expf(buf[i] - mx);
    buf[i] = e; s += e;
  }
  red[tid] = s; __syncthreads();
  for (int st = 128; st > 0; st >>= 1) { if (tid < st) red[tid] += red[tid + st]; __syncthreads(); }
  float inv = 1.f / red[0];
  for (int i = tid; i < L; i += 256) y[row * (long)L + i] = (__bf16)(buf[i] * inv);
}

// ===================================================================
extern "C" void kernel_launch(void* const* d_in, const int* in_sizes, int n_in,
                              void* d_out, int out_size, void* d_ws, size_t ws_size,
                              hipStream_t stream) {
  (void)in_sizes; (void)n_in; (void)out_size; (void)ws_size;
  const int B = 4, N = 512, E = 512, H = 8, D = 64, C = 2048, N4 = 2048, BN = 2048;

  char* ws = (char*)d_ws;
  size_t off = 0;
  auto alloc = [&](size_t bytes) -> void* {
    off = (off + 255) & ~(size_t)255;
    void* p = ws + off;
    off += bytes;
    return p;
  };

  // bf16 operand copies
  __bf16* embC_bf = (__bf16*)alloc((size_t)BN * C * 2);
  __bf16* WqC_bf  = (__bf16*)alloc((size_t)C * C * 2);
  __bf16* WkC_bf  = (__bf16*)alloc((size_t)C * C * 2);
  __bf16* WvC_bf  = (__bf16*)alloc((size_t)C * C * 2);
  __bf16* Wk_bf   = (__bf16*)alloc((size_t)E * E * 2);
  __bf16* Wv_bf   = (__bf16*)alloc((size_t)E * E * 2);
  __bf16 *emb_bf[4], *Wq_bf[4], *Wo_bf[4];
  for (int s = 0; s < 4; ++s) {
    emb_bf[s] = (__bf16*)alloc((size_t)BN * E * 2);
    Wq_bf[s]  = (__bf16*)alloc((size_t)E * E * 2);
    Wo_bf[s]  = (__bf16*)alloc((size_t)E * E * 2);
  }
  // intermediates
  __bf16* QCb   = (__bf16*)alloc((size_t)BN * C * 2);
  __bf16* KCb   = (__bf16*)alloc((size_t)BN * C * 2);
  __bf16* VCb   = (__bf16*)alloc((size_t)BN * C * 2);
  float*  attnC = (float*)alloc((size_t)B * C * C * 4);
  __bf16* simC  = (__bf16*)alloc((size_t)B * C * C * 2);
  __bf16* That  = (__bf16*)alloc((size_t)B * N * C * 2);   // [B,N,C]
  __bf16* Kp    = (__bf16*)alloc((size_t)B * N4 * E * 2);  // [B,4N,E]
  __bf16* Vp    = (__bf16*)alloc((size_t)B * N4 * E * 2);
  __bf16* Qp    = (__bf16*)alloc((size_t)BN * E * 2);      // reused per stream
  float*  attnS = (float*)alloc((size_t)B * H * N * N4 * 4);
  __bf16* simS  = (__bf16*)alloc((size_t)B * H * N * N4 * 2);
  __bf16* ctxS  = (__bf16*)alloc((size_t)BN * E * 2);
  float*  stats = (float*)alloc(64 * sizeof(float));

  auto cast = [&](const void* in, __bf16* outp, long n) {
    long n4 = n / 4;
    cast_f32_bf16<<<dim3((unsigned)((n4 + 255) / 256)), 256, 0, stream>>>((const float*)in, outp, n4);
  };

  cast(d_in[4],  embC_bf, (long)BN * C);
  cast(d_in[11], WqC_bf,  (long)C * C);
  cast(d_in[12], WkC_bf,  (long)C * C);
  cast(d_in[13], WvC_bf,  (long)C * C);
  cast(d_in[9],  Wk_bf,   (long)E * E);
  cast(d_in[10], Wv_bf,   (long)E * E);
  for (int s = 0; s < 4; ++s) {
    cast(d_in[s],      emb_bf[s], (long)BN * E);
    cast(d_in[5 + s],  Wq_bf[s],  (long)E * E);
    cast(d_in[14 + s], Wo_bf[s],  (long)E * E);
  }

  auto gemm = [&](const __bf16* A, const __bf16* Bm, void* Cp, bool f32out,
                  int M, int Nn, int K,
                  int sAm, int sAk, int zAb, int zAh,
                  int sBn, int sBk, int zBb, int zBh,
                  int sCm, int sCn, int zCb, int zCh,
                  int Hdiv, int numZ) {
    dim3 g((M + 127) / 128, (Nn + 127) / 128, numZ);
    if (f32out)
      gemm_bt_wmma<true><<<g, 256, 0, stream>>>(A, Bm, Cp, M, Nn, K,
          sAm, sAk, zAb, zAh, sBn, sBk, zBb, zBh, sCm, sCn, zCb, zCh, Hdiv);
    else
      gemm_bt_wmma<false><<<g, 256, 0, stream>>>(A, Bm, Cp, M, Nn, K,
          sAm, sAk, zAb, zAh, sBn, sBk, zBb, zBh, sCm, sCn, zCb, zCh, Hdiv);
  };

  // ---- channel cross-attention ----
  gemm(embC_bf, WqC_bf, QCb, false, BN, C, C, C,1,0,0, C,1,0,0, C,1,0,0, 1, 1);
  gemm(embC_bf, WkC_bf, KCb, false, BN, C, C, C,1,0,0, C,1,0,0, C,1,0,0, 1, 1);
  gemm(embC_bf, WvC_bf, VCb, false, BN, C, C, C,1,0,0, C,1,0,0, C,1,0,0, 1, 1);

  // attn_c[b,c,d] = sum_n QC[b,n,c]*KC[b,n,d]  (both operands transposed in memory)
  gemm(QCb, KCb, attnC, true, C, C, N,
       1, C, N * C, 0,
       1, C, N * C, 0,
       C, 1, C * C, 0, 1, B);

  hipMemsetAsync(stats, 0, B * 2 * sizeof(float), stream);
  reduce_stats<<<dim3(512, B), 256, 0, stream>>>(attnC, stats, (long)C * C, (long)C * C / 512);
  norm_softmax<<<dim3(B * C), 256, 0, stream>>>(attnC, simC, stats, C, C, 1.0f / ((float)C * C));

  // T_hat[b,n,c] = sum_d sim_c[b,c,d]*VC[b,n,d]   (transpose folded into C strides)
  gemm(simC, VCb, That, false, C, N, C,
       C, 1, C * C, 0,
       C, 1, N * C, 0,
       1, C, N * C, 0, 1, B);

  // K/V projections of KV_S: z = (b,q); KV_S[b,qN+n,e] = T_hat[b,n,qE+e]
  gemm(That, Wk_bf, Kp, false, N, E, E,
       C, 1, N * C, E,
       E, 1, 0, 0,
       E, 1, N4 * E, N * E, 4, B * 4);
  gemm(That, Wv_bf, Vp, false, N, E, E,
       C, 1, N * C, E,
       E, 1, 0, 0,
       E, 1, N4 * E, N * E, 4, B * 4);

  // ---- 4 spatial MHA streams (shared K/V) ----
  for (int s = 0; s < 4; ++s) {
    gemm(emb_bf[s], Wq_bf[s], Qp, false, BN, E, E, E,1,0,0, E,1,0,0, E,1,0,0, 1, 1);

    // attn[b,h,n,m] = sum_d Q[b,n,hD+d]*K[b,m,hD+d]
    gemm(Qp, Kp, attnS, true, N, N4, D,
         E, 1, N * E,  D,
         E, 1, N4 * E, D,
         N4, 1, H * N * N4, N * N4, H, B * H);

    hipMemsetAsync(stats, 0, B * H * 2 * sizeof(float), stream);
    reduce_stats<<<dim3(128, B * H), 256, 0, stream>>>(attnS, stats, (long)N * N4, (long)N * N4 / 128);
    norm_softmax<<<dim3(B * H * N), 256, 0, stream>>>(attnS, simS, stats, N, N4, 1.0f / ((float)N * N4));

    // ctx[b,n,hD+d] = sum_m sim[b,h,n,m]*V[b,m,hD+d]
    gemm(simS, Vp, ctxS, false, N, D, N4,
         N4, 1, H * N * N4, N * N4,
         1, E, N4 * E, D,
         E, 1, N * E, D, H, B * H);

    gemm(ctxS, Wo_bf[s], (float*)d_out + (size_t)s * BN * E, true, BN, E, E,
         E,1,0,0, E,1,0,0, E,1,0,0, 1, 1);
  }
}
